// BiLSTM_CRF_64364379898076
// MI455X (gfx1250) — matrix operand order repaired and checked by reading, not compiled, not run on hardware
//
#include <hip/hip_runtime.h>
#include <hip/hip_bf16.h>

// ---------------------------------------------------------------------------
// BiLSTM-CRF for MI455X (gfx1250): WMMA f16 input-projection GEMMs,
// persistent-workgroup LSTM recurrence (one WGP per direction), wave32
// shuffle-reduced Viterbi.
// ---------------------------------------------------------------------------

typedef __attribute__((ext_vector_type(16))) _Float16 v16h;
typedef __attribute__((ext_vector_type(8)))  float    v8f;

constexpr int TT   = 2048;   // sequence length
constexpr int DD   = 512;    // embed dim
constexpr int H2   = 256;    // per-direction hidden
constexpr int G4   = 4 * H2; // 1024 gate rows
constexpr int NT   = 32;     // tags
constexpr int TAG_START = 30;
constexpr int TAG_END   = 31;
constexpr float NEGV = -10000.0f;

__device__ __forceinline__ float sigm(float x) { return 1.0f / (1.0f + __expf(-x)); }

// ---------------------------------------------------------------------------
// 1) Embedding gather + f32->f16 cast:  xh[t][d] = f16(embed[words[t]][d])
// ---------------------------------------------------------------------------
__global__ void gather_cast_kernel(const int* __restrict__ words,
                                   const float* __restrict__ embed,
                                   _Float16* __restrict__ xh) {
  size_t idx = (size_t)blockIdx.x * blockDim.x + threadIdx.x;
  if (idx >= (size_t)TT * DD) return;
  int t = (int)(idx >> 9);        // / 512
  int d = (int)(idx & 511);
  xh[idx] = (_Float16)embed[(size_t)words[t] * DD + d];
}

// ---------------------------------------------------------------------------
// 2) Generic f32 -> f16 cast (for W_ih matrices)
// ---------------------------------------------------------------------------
__global__ void cast_f16_kernel(const float* __restrict__ src,
                                _Float16* __restrict__ dst, int n) {
  int i = blockIdx.x * blockDim.x + threadIdx.x;
  if (i < n) dst[i] = (_Float16)src[i];
}

// ---------------------------------------------------------------------------
// 3) WMMA GEMM:  Z[t][g] = sum_k X[t'][k] * W[g][k] + b1[g] + b2[g]
//    t' = reverse ? TT-1-t : t.  Block = 256 threads (8 waves), each block
//    computes a 16 x 128 tile of Z; wave w owns the 16x16 tile at col w*16.
//    A/B tiles staged in LDS, K-loop of 512/32 = 16 v_wmma_f32_16x16x32_f16.
// ---------------------------------------------------------------------------
__global__ __launch_bounds__(256) void lstm_pre_gemm(
    const _Float16* __restrict__ X,   // [TT, 512] f16
    const _Float16* __restrict__ W,   // [1024, 512] f16 (row-major, gate rows)
    const float* __restrict__ b1, const float* __restrict__ b2,
    float* __restrict__ Z,            // [TT, 1024] f32
    int reverse) {
  __shared__ _Float16 As[16 * 32];     // A tile: 16 t-rows x 32 k
  __shared__ _Float16 Bs[128 * 32];    // W tile: 128 gate rows x 32 k

  const int tid  = threadIdx.x;
  const int lane = tid & 31;
  const int wave = tid >> 5;
  const int t0   = blockIdx.x * 16;
  const int g0   = blockIdx.y * 128;
  const int hi   = lane >> 4;          // lane >= 16 ?
  const int N    = lane & 15;          // also M for A-fragment

  v8f acc = {};

  for (int kk = 0; kk < DD; kk += 32) {
    // --- stage A tile: 512 f16, 2 per thread -----------------------------
    {
      int i = tid * 2;
      int r = i >> 5, c = i & 31;
      int trow = reverse ? (TT - 1 - (t0 + r)) : (t0 + r);
      const _Float16* xp = X + (size_t)trow * DD + kk + c;
      As[i]     = xp[0];
      As[i + 1] = xp[1];
    }
    // --- stage W tile: 128 rows x 32 k, 16 contiguous f16 per thread -----
    {
      int row = tid >> 1;
      int c0  = (tid & 1) * 16;
      const _Float16* wp = W + (size_t)(g0 + row) * DD + kk + c0;
#pragma unroll
      for (int j = 0; j < 16; ++j) Bs[row * 32 + c0 + j] = wp[j];
      if (kk + 32 < DD) __builtin_prefetch(wp + 32, 0, 1);  // global_prefetch_b8
    }
    __syncthreads();

    // --- build fragments per ISA 7.12.2 (wave32) -------------------------
    v16h afrag, bfrag;
    {
      const int ab = hi ? 8 : 0;             // K base for elems 0..7
#pragma unroll
      for (int j = 0; j < 8; ++j) {
        afrag[j]     = As[N * 32 + ab + j];        // K = ab + j
        afrag[8 + j] = As[N * 32 + ab + 16 + j];   // K = ab + 16 + j
      }
      const int kb = hi ? 16 : 0;            // unused; B fragment uses A's K base ab
      const int gr = wave * 16 + N;          // gate row for this lane's column
#pragma unroll
      for (int j = 0; j < 8; ++j) {
        bfrag[j]     = Bs[gr * 32 + ab + j];
        bfrag[8 + j] = Bs[gr * 32 + ab + 16 + j];
      }
    }

    acc = __builtin_amdgcn_wmma_f32_16x16x32_f16(
        /*neg_a=*/false, afrag, /*neg_b=*/false, bfrag,
        /*c_mod=*/(short)0, acc, /*reuse_a=*/false, /*reuse_b=*/false);
    __syncthreads();
  }

  // --- epilogue: D elem r -> row M = r + hi*8, col N -----------------------
  const int g = g0 + wave * 16 + N;
  const float bias = b1[g] + b2[g];
#pragma unroll
  for (int r = 0; r < 8; ++r) {
    int row = t0 + r + hi * 8;
    Z[(size_t)row * G4 + g] = acc[r] + bias;
  }
}

// ---------------------------------------------------------------------------
// 4) Persistent recurrence: one 1024-thread block per direction (grid=2).
//    Thread g computes gate row g: z_g = Zpre[t][g] + dot(W_hh[g,:], h).
//    h (256 f32) lives in LDS; c lives in registers of threads 0..255.
// ---------------------------------------------------------------------------
__global__ __launch_bounds__(1024) void lstm_scan_kernel(
    const float* __restrict__ zf, const float* __restrict__ zb,
    const float* __restrict__ whf, const float* __restrict__ whb,
    const float* __restrict__ h0, const float* __restrict__ c0,
    float* __restrict__ hf_out, float* __restrict__ hb_out) {
  const int dir = blockIdx.x;                 // 0 = forward, 1 = backward
  const float* Z   = dir ? zb : zf;           // [TT, 1024]
  const float* Wh  = dir ? whb : whf;         // [1024, 256]
  float* hout      = dir ? hb_out : hf_out;   // [TT, 256]

  __shared__ float hbuf[H2];
  __shared__ float zbuf[G4];

  const int g = threadIdx.x;
  float c = 0.0f;
  if (g < H2) {
    c = c0[dir * H2 + g];
    hbuf[g] = h0[dir * H2 + g];
  }
  __syncthreads();

  const float4* wrow4 = (const float4*)(Wh + (size_t)g * H2);

  for (int t = 0; t < TT; ++t) {
    float acc = Z[(size_t)t * G4 + g];
#pragma unroll 8
    for (int k = 0; k < H2 / 4; ++k) {
      float4 w = wrow4[k];
      acc += w.x * hbuf[4 * k] + w.y * hbuf[4 * k + 1] +
             w.z * hbuf[4 * k + 2] + w.w * hbuf[4 * k + 3];
    }
    zbuf[g] = acc;
    __syncthreads();

    float hreg = 0.0f;
    if (g < H2) {
      float zi = zbuf[g], zfo = zbuf[H2 + g], zg = zbuf[2 * H2 + g],
            zo = zbuf[3 * H2 + g];
      c    = sigm(zfo) * c + sigm(zi) * tanhf(zg);
      hreg = sigm(zo) * tanhf(c);
    }
    __syncthreads();                          // hbuf reads done before overwrite
    if (g < H2) {
      hbuf[g] = hreg;
      int ot = dir ? (TT - 1 - t) : t;        // backward outputs un-reversed
      hout[(size_t)ot * H2 + g] = hreg;
    }
    __syncthreads();
  }
}

// ---------------------------------------------------------------------------
// 5) Tag projection: frames[t][tag] = [hf|hb][t] . w_tag[tag] + b_tag[tag]
// ---------------------------------------------------------------------------
__global__ void tag_proj_kernel(const float* __restrict__ hf,
                                const float* __restrict__ hb,
                                const float* __restrict__ wt,
                                const float* __restrict__ bt,
                                float* __restrict__ frames) {
  int idx = blockIdx.x * blockDim.x + threadIdx.x;   // t*32 + tag
  if (idx >= TT * NT) return;
  int t = idx >> 5, tag = idx & 31;
  const float* w = wt + (size_t)tag * (2 * H2);
  const float* hfp = hf + (size_t)t * H2;
  const float* hbp = hb + (size_t)t * H2;
  float acc = bt[tag];
#pragma unroll 4
  for (int k = 0; k < H2; ++k) acc += hfp[k] * w[k];
#pragma unroll 4
  for (int k = 0; k < H2; ++k) acc += hbp[k] * w[H2 + k];
  frames[idx] = acc;
}

// ---------------------------------------------------------------------------
// 6) Viterbi / CRF scan. Block = (32,32): lane = 'from', y = 'to'.
//    Each wave reduces over 'from' with __shfl_xor butterflies (max/argmax
//    with first-occurrence tie-break, then sum-exp) -> alpha, backpointers.
//    Then terminal logsumexp/argmax and a serial backtrace by thread 0.
// ---------------------------------------------------------------------------
__global__ __launch_bounds__(1024) void viterbi_kernel(
    const float* __restrict__ frames, const float* __restrict__ trans,
    int* __restrict__ bp, float* __restrict__ out) {
  __shared__ float tr[NT * NT];
  __shared__ float alpha[NT];
  __shared__ float alpha_new[NT];
  __shared__ float score_s;
  __shared__ int   best_s;

  const int from = threadIdx.x;
  const int to   = threadIdx.y;
  const int tid  = to * 32 + from;

  tr[tid] = trans[tid];                       // 32x32 = 1024 = blockDim
  if (tid < NT) alpha[tid] = (tid == TAG_START) ? 0.0f : NEGV;
  __syncthreads();

  for (int t = 0; t < TT; ++t) {
    float s = alpha[from] + frames[t * NT + to] + tr[from * NT + to];

    // wave32 max + argmax (lowest index on ties, matching jnp.argmax)
    float m = s; int arg = from;
#pragma unroll
    for (int off = 16; off > 0; off >>= 1) {
      float om = __shfl_xor(m, off, 32);
      int   oa = __shfl_xor(arg, off, 32);
      if (om > m || (om == m && oa < arg)) { m = om; arg = oa; }
    }
    // wave32 logsumexp
    float e = __expf(s - m);
#pragma unroll
    for (int off = 16; off > 0; off >>= 1) e += __shfl_xor(e, off, 32);

    if (from == 0) {
      alpha_new[to] = m + __logf(e);
      bp[t * NT + to] = arg;
    }
    __syncthreads();
    if (tid < NT) alpha[tid] = alpha_new[tid];
    __syncthreads();
  }

  // terminal: term = alpha + trans[:, END]
  if (to == 0) {
    float term = alpha[from] + tr[from * NT + TAG_END];
    float m = term; int arg = from;
#pragma unroll
    for (int off = 16; off > 0; off >>= 1) {
      float om = __shfl_xor(m, off, 32);
      int   oa = __shfl_xor(arg, off, 32);
      if (om > m || (om == m && oa < arg)) { m = om; arg = oa; }
    }
    float e = __expf(term - m);
#pragma unroll
    for (int off = 16; off > 0; off >>= 1) e += __shfl_xor(e, off, 32);
    if (from == 0) { score_s = m + __logf(e); best_s = arg; }
  }
  __syncthreads();

  if (tid == 0) {
    out[0] = score_s;
    float* path = out + 1;                    // path emitted as float values
    int tag = best_s;
    path[TT - 1] = (float)tag;
    for (int t = TT - 1; t >= 1; --t) {
      tag = bp[t * NT + tag];
      path[t - 1] = (float)tag;
    }
  }
}

// ---------------------------------------------------------------------------
// Host-side launch (graph-capture safe: kernels only, all on `stream`)
// ---------------------------------------------------------------------------
extern "C" void kernel_launch(void* const* d_in, const int* in_sizes, int n_in,
                              void* d_out, int out_size, void* d_ws, size_t ws_size,
                              hipStream_t stream) {
  const int*   words  = (const int*)d_in[0];
  const float* embed  = (const float*)d_in[1];
  const float* w_ih_f = (const float*)d_in[2];
  const float* w_hh_f = (const float*)d_in[3];
  const float* b_ih_f = (const float*)d_in[4];
  const float* b_hh_f = (const float*)d_in[5];
  const float* w_ih_b = (const float*)d_in[6];
  const float* w_hh_b = (const float*)d_in[7];
  const float* b_ih_b = (const float*)d_in[8];
  const float* b_hh_b = (const float*)d_in[9];
  const float* h0     = (const float*)d_in[10];
  const float* c0     = (const float*)d_in[11];
  const float* w_tag  = (const float*)d_in[12];
  const float* b_tag  = (const float*)d_in[13];
  const float* trans  = (const float*)d_in[14];
  float* out = (float*)d_out;

  // workspace carve-up (~24.5 MB total)
  char* ws = (char*)d_ws;
  _Float16* xh  = (_Float16*)ws;                          //  2 MB  [TT,512] f16
  _Float16* wfh = xh  + (size_t)TT * DD;                  //  1 MB  [1024,512] f16
  _Float16* wbh = wfh + (size_t)G4 * DD;                  //  1 MB
  float* zf     = (float*)(wbh + (size_t)G4 * DD);        //  8 MB  [TT,1024] f32
  float* zb     = zf + (size_t)TT * G4;                   //  8 MB
  float* hf     = zb + (size_t)TT * G4;                   //  2 MB  [TT,256]
  float* hb     = hf + (size_t)TT * H2;                   //  2 MB
  float* frames = hb + (size_t)TT * H2;                   // 256 KB [TT,32]
  int*   bp     = (int*)(frames + (size_t)TT * NT);       // 256 KB [TT,32]

  gather_cast_kernel<<<(TT * DD) / 256, 256, 0, stream>>>(words, embed, xh);
  cast_f16_kernel<<<(G4 * DD) / 256, 256, 0, stream>>>(w_ih_f, wfh, G4 * DD);
  cast_f16_kernel<<<(G4 * DD) / 256, 256, 0, stream>>>(w_ih_b, wbh, G4 * DD);

  dim3 gg(TT / 16, G4 / 128);
  lstm_pre_gemm<<<gg, 256, 0, stream>>>(xh, wfh, b_ih_f, b_hh_f, zf, 0);
  lstm_pre_gemm<<<gg, 256, 0, stream>>>(xh, wbh, b_ih_b, b_hh_b, zb, 1);

  lstm_scan_kernel<<<2, 1024, 0, stream>>>(zf, zb, w_hh_f, w_hh_b, h0, c0, hf, hb);

  tag_proj_kernel<<<(TT * NT) / 256, 256, 0, stream>>>(hf, hb, w_tag, b_tag, frames);

  viterbi_kernel<<<dim3(1, 1, 1), dim3(32, 32, 1), 0, stream>>>(frames, trans, bp, out);
}
